// SynthNNAdapter_32427003085099
// MI455X (gfx1250) — matrix operand order repaired and checked by reading, not compiled, not run on hardware
//
#include <hip/hip_runtime.h>
#include <hip/hip_bf16.h>

typedef __attribute__((ext_vector_type(16))) __bf16 v16bf;
typedef __attribute__((ext_vector_type(8)))  float  v8f;

#define HID   4096
#define NN    32
#define MTOT  8192     // B*S
#define KCH1  128      // HID/32 k-chunks for GEMM1
#define NT2   256      // HID/16 n-tiles for GEMM2

// ---------------------------------------------------------------------------
// A-fragment loader: 16x32 bf16 tile from an fp32 row-major matrix.
// ISA layout (CDNA5 ISA, 16-bit A 16x32): lane l holds row M=l&15;
// lanes>=16 shift K by +8; elements e>=8 shift K by +16.
// => per lane: four float4 loads at kbase+{0,4,16,20}, kbase = chunk*32 + koff.
// ---------------------------------------------------------------------------
__device__ inline v16bf load_a_frag(const float* __restrict__ rowptr, int kbase) {
    float4 f0 = *(const float4*)(rowptr + kbase);
    float4 f1 = *(const float4*)(rowptr + kbase + 4);
    float4 f2 = *(const float4*)(rowptr + kbase + 16);
    float4 f3 = *(const float4*)(rowptr + kbase + 20);
    v16bf a;
    a[0]  = (__bf16)f0.x; a[1]  = (__bf16)f0.y; a[2]  = (__bf16)f0.z; a[3]  = (__bf16)f0.w;
    a[4]  = (__bf16)f1.x; a[5]  = (__bf16)f1.y; a[6]  = (__bf16)f1.z; a[7]  = (__bf16)f1.w;
    a[8]  = (__bf16)f2.x; a[9]  = (__bf16)f2.y; a[10] = (__bf16)f2.z; a[11] = (__bf16)f2.w;
    a[12] = (__bf16)f3.x; a[13] = (__bf16)f3.y; a[14] = (__bf16)f3.z; a[15] = (__bf16)f3.w;
    return a;
}

// ---------------------------------------------------------------------------
// Kernel 1: pre-swizzle W1 (4096x32) and W2 (64x4096) into bf16 WMMA
// B-fragment order: frag[chunk/tile][half/k][lane][e], where per ISA B layout
// K = 32*chunk + (lane>=16 ? 8 : 0) + (e & 7) + (e>=8 ? 16 : 0),  N = base + (lane & 15).
// ---------------------------------------------------------------------------
__global__ __launch_bounds__(256) void prep_weights_k(
    const float* __restrict__ W1, const float* __restrict__ W2,
    __bf16* __restrict__ w1f, __bf16* __restrict__ w2f) {
    const int W1F = KCH1 * 2 * 32 * 16;   // 131072
    const int W2F = NT2  * 2 * 32 * 16;   // 262144
    int idx = blockIdx.x * blockDim.x + threadIdx.x;
    if (idx < W1F) {
        int e = idx & 15, l = (idx >> 4) & 31, h = (idx >> 9) & 1, c = idx >> 10;
        int K = c * 32 + ((l >= 16) ? 8 : 0) + (e & 7) + ((e >= 8) ? 16 : 0);
        int N = h * 16 + (l & 15);
        w1f[idx] = (__bf16)W1[K * NN + N];
    } else if (idx < W1F + W2F) {
        int j = idx - W1F;
        int e = j & 15, l = (j >> 4) & 31, k = (j >> 9) & 1, t = j >> 10;
        int K = k * 32 + ((l >= 16) ? 8 : 0) + (e & 7) + ((e >= 8) ? 16 : 0);
        int N = t * 16 + (l & 15);
        w2f[j] = (__bf16)W2[K * HID + N];
    }
}

// ---------------------------------------------------------------------------
// Kernel 2: ext = tanh(x @ W1 + b1).  One wave per 16-row tile; two WMMA
// accumulators cover N=0..15 and N=16..31.  K-loop: 128 chunks of 32.
// ---------------------------------------------------------------------------
__global__ __launch_bounds__(256) void gemm1_tanh_k(
    const float* __restrict__ x, const __bf16* __restrict__ w1f,
    const float* __restrict__ b1, float* __restrict__ ext) {
    int lane = threadIdx.x & 31;
    int wave = threadIdx.x >> 5;
    int mt   = blockIdx.x * 8 + wave;                 // 0..511 row tiles
    int koff = (lane >= 16) ? 8 : 0;
    const float* rowptr = x + (long)(mt * 16 + (lane & 15)) * HID;

    v8f acc0 = {}; v8f acc1 = {};
    #pragma unroll 4
    for (int c = 0; c < KCH1; ++c) {
        v16bf a  = load_a_frag(rowptr, c * 32 + koff);
        v16bf b0 = *(const v16bf*)(w1f + ((c * 2 + 0) * 32 + lane) * 16);
        v16bf bx = *(const v16bf*)(w1f + ((c * 2 + 1) * 32 + lane) * 16);
        acc0 = __builtin_amdgcn_wmma_f32_16x16x32_bf16(false, a, false, b0, (short)0, acc0, false, false);
        acc1 = __builtin_amdgcn_wmma_f32_16x16x32_bf16(false, a, false, bx, (short)0, acc1, false, false);
    }
    // C/D layout: lane l, vgpr r -> M = r + (l>=16 ? 8 : 0), N = l & 15.
    int n0    = lane & 15;
    float bs0 = b1[n0], bs1 = b1[16 + n0];
    int mrow  = mt * 16 + ((lane >= 16) ? 8 : 0);
    #pragma unroll
    for (int r = 0; r < 8; ++r) {
        int m = mrow + r;
        ext[m * NN + n0]      = tanhf(acc0[r] + bs0);
        ext[m * NN + 16 + n0] = tanhf(acc1[r] + bs1);
    }
}

// ---------------------------------------------------------------------------
// Kernel 3: the sequential oscillator scan.  One wave; lane = node.
// Wc is a nearest-neighbor chain (weight 0.5) => coupling via two lane
// shuffles (ds_bpermute), no matmul.  81,920 dependent steps total.
// ---------------------------------------------------------------------------
__global__ __launch_bounds__(32) void osc_scan_k(
    const float* __restrict__ ext, const float* __restrict__ init_phase,
    float* __restrict__ states) {
    int lane = threadIdx.x;
    float p = init_phase[lane];
    float a = 0.1f;
    const float omega = 6.28318530717958647692f * (1.0f + (float)lane * (0.5f / 32.0f));
    for (int m = 0; m < MTOT; ++m) {
        float e = ext[m * NN + lane];
        #pragma unroll
        for (int s = 0; s < 10; ++s) {
            float sv = a * __sinf(p);
            float ls = __shfl(sv, lane - 1, 32);
            float rs = __shfl(sv, lane + 1, 32);
            if (lane == 0)  ls = 0.0f;
            if (lane == 31) rs = 0.0f;
            float coup = 0.5f * (ls + rs);
            p += 0.01f * (omega + coup + e);
            a += 0.01f * e * __cosf(p);
            a = fminf(fmaxf(a, 0.0f), 1.0f);
        }
        states[m * 64 + lane]      = p;
        states[m * 64 + 32 + lane] = a;
    }
}

// ---------------------------------------------------------------------------
// Kernel 4: out = x + states @ W2 + b2.  One wave per (16-row tile x 16
// N-tiles); A fragments (K=64 -> 2 chunks) loaded once per wave.
// ---------------------------------------------------------------------------
__global__ __launch_bounds__(256) void gemm2_res_k(
    const float* __restrict__ states, const __bf16* __restrict__ w2f,
    const float* __restrict__ x, const float* __restrict__ b2,
    float* __restrict__ out) {
    int lane = threadIdx.x & 31;
    int wave = threadIdx.x >> 5;
    int gw   = blockIdx.x * 8 + wave;   // 0..8191
    int mt   = gw >> 4;                 // 0..511 row tile
    int ng   = gw & 15;                 // group of 16 N-tiles
    int koff = (lane >= 16) ? 8 : 0;
    const float* srow = states + (long)(mt * 16 + (lane & 15)) * 64;
    v16bf a0 = load_a_frag(srow, 0  + koff);   // K = 0..31
    v16bf a1 = load_a_frag(srow, 32 + koff);   // K = 32..63
    int mrow = mt * 16 + ((lane >= 16) ? 8 : 0);

    for (int j = 0; j < 16; ++j) {
        int nt = ng * 16 + j;
        v16bf b0 = *(const v16bf*)(w2f + ((nt * 2 + 0) * 32 + lane) * 16);
        v16bf bx = *(const v16bf*)(w2f + ((nt * 2 + 1) * 32 + lane) * 16);
        v8f acc = {};
        acc = __builtin_amdgcn_wmma_f32_16x16x32_bf16(false, a0, false, b0, (short)0, acc, false, false);
        acc = __builtin_amdgcn_wmma_f32_16x16x32_bf16(false, a1, false, bx, (short)0, acc, false, false);
        int col = nt * 16 + (lane & 15);
        float bias = b2[col];
        #pragma unroll
        for (int r = 0; r < 8; ++r) {
            long idx = (long)(mrow + r) * HID + col;
            out[idx] = x[idx] + acc[r] + bias;
        }
    }
}

// ---------------------------------------------------------------------------
// Launch.  Workspace layout (needs ~3.75 MB):
//   [0, 256K)        W1 bf16 fragments
//   [256K, 768K)     W2 bf16 fragments
//   [768K, 1.75M)    ext  fp32 [8192 x 32]
//   [1.75M, 3.75M)   states fp32 [8192 x 64]
// ---------------------------------------------------------------------------
extern "C" void kernel_launch(void* const* d_in, const int* in_sizes, int n_in,
                              void* d_out, int out_size, void* d_ws, size_t ws_size,
                              hipStream_t stream) {
    const float* x  = (const float*)d_in[0];
    const float* W1 = (const float*)d_in[1];
    const float* b1 = (const float*)d_in[2];
    const float* W2 = (const float*)d_in[3];
    const float* b2 = (const float*)d_in[4];
    const float* ip = (const float*)d_in[5];
    float* out = (float*)d_out;

    char* ws = (char*)d_ws;
    __bf16* w1f = (__bf16*)(ws);
    __bf16* w2f = (__bf16*)(ws + 262144);
    float*  ext = (float*) (ws + 262144 + 524288);
    float*  st  = (float*) (ws + 262144 + 524288 + 1048576);

    prep_weights_k<<<1536, 256, 0, stream>>>(W1, W2, w1f, w2f);
    gemm1_tanh_k  <<<64,   256, 0, stream>>>(x, w1f, b1, ext);
    osc_scan_k    <<<1,    32,  0, stream>>>(ext, ip, st);
    gemm2_res_k   <<<1024, 256, 0, stream>>>(st, w2f, x, b2, out);
}
